// Attention1d_6786048328224
// MI455X (gfx1250) — matrix operand order repaired and checked by reading, not compiled, not run on hardware
//
#include <hip/hip_runtime.h>
#include <hip/hip_bf16.h>

// ---------------------------------------------------------------------------
// Attention1d on MI455X (gfx1250): bf16 WMMA pipeline
//   x[4,512,2048] --(Wqkv)--> q,k,v[4,8,2048,64] --flash-attn--> [4,2048,512]
//   --(Wout + bias)--> out[4,512,2048] (fp32)
// ---------------------------------------------------------------------------

#define B_    4
#define C_    512
#define L_    2048
#define H_    8
#define D_    64
#define O3_   1536
#define SCALE 0.125f   // 64^-0.5

typedef __attribute__((ext_vector_type(16))) __bf16 v16bf;
typedef __attribute__((ext_vector_type(8)))  float  v8f;

union frag_u { uint4 u[2]; v16bf v; };

// ---- gfx1250 async global->LDS path (guarded; falls back to VGPR bounce) ---
#if defined(__HIP_DEVICE_COMPILE__) &&                                        \
    __has_builtin(__builtin_amdgcn_global_load_async_to_lds_b128) &&          \
    __has_builtin(__builtin_amdgcn_s_wait_asynccnt)
#define USE_ASYNC_LDS 1
#define AS(n) __attribute__((address_space(n)))
typedef __attribute__((__vector_size__(4 * sizeof(int)))) int vi4;
__device__ __forceinline__ void async_g2l_b128(const void* g, void* l) {
  __builtin_amdgcn_global_load_async_to_lds_b128(
      (AS(1) vi4*)(void*)g, (AS(3) vi4*)l, 0, 0);
}
#endif

// ---- WMMA wrapper ----------------------------------------------------------
__device__ __forceinline__ v8f wmma_bf16(v16bf a, v16bf b, v8f c) {
  return __builtin_amdgcn_wmma_f32_16x16x32_bf16(false, a, false, b,
                                                 (short)0, c, false, false);
}

// ---- Fragment loaders (contiguous-k lane layouts, same k order in A and B) ----------------------------
// A (16x32, row-major src, element (m,k) at p[m*ld+k]).
// lane<16: M=lane, K=0..15; lane>=16: same M, K=16..31. Contiguous 32B.
__device__ __forceinline__ v16bf load_a(const __bf16* p, int ld) {
  int lane = threadIdx.x & 31;
  int m  = lane & 15;
  int kb = (lane & 16) ? 16 : 0;
  const uint4* q = reinterpret_cast<const uint4*>(p + m * ld + kb);
  frag_u r;
  r.u[0] = q[0];        // K = kb .. kb+7
  r.u[1] = q[1];        // K = kb+8 .. kb+15
  return r.v;
}

// B (32x16) when memory holds B^T row-major: element (k,n) at p[n*ld+k].
// lane<16: N=lane, K=0..15; lane>=16: N=lane-16, K=16..31. Contiguous 32B.
__device__ __forceinline__ v16bf load_bt(const __bf16* p, int ld) {
  int lane = threadIdx.x & 31;
  int n  = lane & 15;
  int kb = (lane & 16) ? 16 : 0;
  const uint4* q = reinterpret_cast<const uint4*>(p + n * ld + kb);
  frag_u r;
  r.u[0] = q[0];
  r.u[1] = q[1];
  return r.v;
}

// B (32x16), row-major src: element (k,n) at p[k*ld+n]. Strided gather.
__device__ __forceinline__ v16bf load_b_strided(const __bf16* p, int ld) {
  int lane = threadIdx.x & 31;
  int n  = lane & 15;
  int kb = (lane & 16) ? 16 : 0;
  v16bf r;
#pragma unroll
  for (int e = 0; e < 16; ++e) r[e] = p[(kb + e) * ld + n];
  return r;
}

// ---- K0a: convert weights to bf16 (fold softmax scale into Q rows) ---------
__global__ __launch_bounds__(256) void convert_weights(
    const float* __restrict__ wqkv, const float* __restrict__ wout,
    __bf16* __restrict__ wq_b, __bf16* __restrict__ wo_b) {
  int i = blockIdx.x * blockDim.x + threadIdx.x;
  if (i < O3_ * C_) {
    float v = wqkv[i];
    if (i < C_ * C_) v *= SCALE;         // rows o<512 are Q
    wq_b[i] = (__bf16)v;
  }
  if (i < C_ * C_) wo_b[i] = (__bf16)wout[i];
}

// ---- K0b: x[b,c,l] -> xT[b,l,c] bf16 ---------------------------------------
__global__ __launch_bounds__(256) void convert_xT(
    const float* __restrict__ x, __bf16* __restrict__ xT) {
  int i = blockIdx.x * blockDim.x + threadIdx.x;
  if (i >= B_ * C_ * L_) return;
  int l = i % L_;
  int c = (i / L_) % C_;
  int b = i / (L_ * C_);
  xT[((size_t)b * L_ + l) * C_ + c] = (__bf16)x[i];
}

// ---- K1: QKV GEMM  qkv[o,l] = Wqkv[o,:] . x[:,l]  -> q/k/v[b,h,l,d] --------
// wave computes 32(o) x 64(l); grid (L/64, O3/(32*8), B), block 256 (8 waves)
__global__ __launch_bounds__(256) void qkv_gemm(
    const __bf16* __restrict__ wq, const __bf16* __restrict__ xT,
    __bf16* __restrict__ qs, __bf16* __restrict__ ks, __bf16* __restrict__ vs) {
  int wave = threadIdx.x >> 5;
  int b  = blockIdx.z;
  int o0 = (blockIdx.y * 8 + wave) * 32;
  int l0 = blockIdx.x * 64;

  v8f acc[2][4];
#pragma unroll
  for (int ms = 0; ms < 2; ++ms)
#pragma unroll
    for (int f = 0; f < 4; ++f)
#pragma unroll
      for (int e = 0; e < 8; ++e) acc[ms][f][e] = 0.f;

  const __bf16* xb = xT + (size_t)b * L_ * C_;
  for (int kc = 0; kc < C_; kc += 32) {
    __builtin_prefetch(wq + (size_t)o0 * C_ + kc + 64, 0, 3);
    v16bf a0 = load_a(wq + (size_t)(o0)      * C_ + kc, C_);
    v16bf a1 = load_a(wq + (size_t)(o0 + 16) * C_ + kc, C_);
#pragma unroll
    for (int f = 0; f < 4; ++f) {
      v16bf bf = load_bt(xb + (size_t)(l0 + 16 * f) * C_ + kc, C_);
      acc[0][f] = wmma_bf16(a0, bf, acc[0][f]);
      acc[1][f] = wmma_bf16(a1, bf, acc[1][f]);
    }
  }

  // scatter epilogue into [b,h,l,d] bf16
  int lane = threadIdx.x & 31;
  int hl = lane >> 4, n = lane & 15;
  int part  = o0 / C_;                  // 0=q 1=k 2=v (32-row tile stays in one part)
  int obase = o0 % C_;
  __bf16* dst = (part == 0) ? qs : (part == 1) ? ks : vs;
#pragma unroll
  for (int ms = 0; ms < 2; ++ms) {
#pragma unroll
    for (int f = 0; f < 4; ++f) {
      int l = l0 + 16 * f + n;
#pragma unroll
      for (int r = 0; r < 8; ++r) {
        int o = obase + 16 * ms + r + 8 * hl;
        int h = o >> 6, d = o & 63;
        dst[((size_t)(b * H_ + h) * L_ + l) * D_ + d] = (__bf16)acc[ms][f][r];
      }
    }
  }
}

// ---- K2: fused flash attention --------------------------------------------
// block = 128 threads (4 waves); wave owns 16 query rows; block covers 64.
// grid (L/64, B*H)
__global__ __launch_bounds__(128) void attn_kernel(
    const __bf16* __restrict__ qs, const __bf16* __restrict__ ks,
    const __bf16* __restrict__ vs, __bf16* __restrict__ attnT) {
  __shared__ alignas(16) __bf16 sK[64 * 64];
  __shared__ alignas(16) __bf16 sV[64 * 64];
  __shared__ alignas(16) __bf16 sP[4][16 * 32];

  int bh = blockIdx.y;
  int b = bh >> 3, h = bh & 7;
  int wave = threadIdx.x >> 5;
  int lane = threadIdx.x & 31;
  int hl = lane >> 4, n = lane & 15;
  int i0 = blockIdx.x * 64 + wave * 16;

  // Q fragments (rows i0..i0+15, K = d)
  const __bf16* qbase = qs + ((size_t)bh * L_ + i0) * D_;
  v16bf qa0 = load_a(qbase, D_);        // d 0..31
  v16bf qa1 = load_a(qbase + 32, D_);   // d 32..63

  v8f acc[4];
  float mrow[8], lrow[8];
#pragma unroll
  for (int f = 0; f < 4; ++f)
#pragma unroll
    for (int e = 0; e < 8; ++e) acc[f][e] = 0.f;
#pragma unroll
  for (int r = 0; r < 8; ++r) { mrow[r] = -1e30f; lrow[r] = 0.f; }

  for (int j0 = 0; j0 < L_; j0 += 64) {
    // cooperative K/V chunk load: [64 rows j][64 d] bf16 = 8KB each
    {
      const uint4* gk = reinterpret_cast<const uint4*>(ks + ((size_t)bh * L_ + j0) * D_);
      const uint4* gv = reinterpret_cast<const uint4*>(vs + ((size_t)bh * L_ + j0) * D_);
      uint4* lk = reinterpret_cast<uint4*>(sK);
      uint4* lv = reinterpret_cast<uint4*>(sV);
#ifdef USE_ASYNC_LDS
      for (int u = threadIdx.x; u < 512; u += 128) {
        async_g2l_b128(gk + u, lk + u);
        async_g2l_b128(gv + u, lv + u);
      }
      __builtin_amdgcn_s_wait_asynccnt(0);
#else
      for (int u = threadIdx.x; u < 512; u += 128) { lk[u] = gk[u]; lv[u] = gv[u]; }
#endif
    }
    __syncthreads();

#pragma unroll
    for (int jj = 0; jj < 2; ++jj) {        // two 32-wide column chunks
      // --- S = Q^T K for two 16-col subtiles ---
      v8f s0, s1;
#pragma unroll
      for (int e = 0; e < 8; ++e) { s0[e] = 0.f; s1[e] = 0.f; }
      const __bf16* kb = sK + (jj * 32) * D_;
      v16bf k00 = load_bt(kb, D_);                 // cols j..j+15, d 0..31
      v16bf k01 = load_bt(kb + 32, D_);            // d 32..63
      s0 = wmma_bf16(qa0, k00, s0);
      s0 = wmma_bf16(qa1, k01, s0);
      v16bf k10 = load_bt(kb + 16 * D_, D_);       // cols j+16..j+31
      v16bf k11 = load_bt(kb + 16 * D_ + 32, D_);
      s1 = wmma_bf16(qa0, k10, s1);
      s1 = wmma_bf16(qa1, k11, s1);

      // --- online softmax update (rows r+8*hl handled per lane) ---
      float tmax[8];
#pragma unroll
      for (int r = 0; r < 8; ++r) tmax[r] = fmaxf(s0[r], s1[r]);
#pragma unroll
      for (int msk = 1; msk < 16; msk <<= 1)
#pragma unroll
        for (int r = 0; r < 8; ++r)
          tmax[r] = fmaxf(tmax[r], __shfl_xor(tmax[r], msk, 32));

      float alpha[8];
#pragma unroll
      for (int r = 0; r < 8; ++r) {
        float mn = fmaxf(mrow[r], tmax[r]);
        alpha[r] = __expf(mrow[r] - mn);
        mrow[r] = mn;
      }
#pragma unroll
      for (int r = 0; r < 8; ++r) {
        s0[r] = __expf(s0[r] - mrow[r]);
        s1[r] = __expf(s1[r] - mrow[r]);
      }
      float rsum[8];
#pragma unroll
      for (int r = 0; r < 8; ++r) rsum[r] = s0[r] + s1[r];
#pragma unroll
      for (int msk = 1; msk < 16; msk <<= 1)
#pragma unroll
        for (int r = 0; r < 8; ++r)
          rsum[r] += __shfl_xor(rsum[r], msk, 32);
#pragma unroll
      for (int r = 0; r < 8; ++r) lrow[r] = lrow[r] * alpha[r] + rsum[r];
#pragma unroll
      for (int f = 0; f < 4; ++f)
#pragma unroll
        for (int r = 0; r < 8; ++r) acc[f][r] *= alpha[r];

      // --- P (C-layout) -> LDS -> A-layout fragment ---
      __bf16* pw = sP[wave];
#pragma unroll
      for (int r = 0; r < 8; ++r) {
        int row = r + 8 * hl;
        pw[row * 32 + n]      = (__bf16)s0[r];
        pw[row * 32 + 16 + n] = (__bf16)s1[r];
      }
      v16bf pa = load_a(pw, 32);

      // --- O += P x V (K = 32 local cols) ---
      const __bf16* vb = sV + (jj * 32) * D_;
#pragma unroll
      for (int f = 0; f < 4; ++f) {
        v16bf vf = load_b_strided(vb + 16 * f, D_);
        acc[f] = wmma_bf16(pa, vf, acc[f]);
      }
    }
    __syncthreads();
  }

  // epilogue: normalize and store into attnT[b,l,c] bf16 (c = h*64+d)
#pragma unroll
  for (int f = 0; f < 4; ++f) {
#pragma unroll
    for (int r = 0; r < 8; ++r) {
      int i = i0 + r + 8 * hl;
      int d = 16 * f + n;
      float v = acc[f][r] / lrow[r];
      attnT[((size_t)b * L_ + i) * C_ + h * D_ + d] = (__bf16)v;
    }
  }
}

// ---- K3: output projection + bias -----------------------------------------
// wave computes 32(o) x 64(l); grid (L/64, C/(32*8), B), block 256
__global__ __launch_bounds__(256) void out_gemm(
    const __bf16* __restrict__ wo, const __bf16* __restrict__ attnT,
    const float* __restrict__ bias, float* __restrict__ out) {
  int wave = threadIdx.x >> 5;
  int lane = threadIdx.x & 31;
  int hl = lane >> 4, n = lane & 15;
  int b  = blockIdx.z;
  int o0 = (blockIdx.y * 8 + wave) * 32;
  int l0 = blockIdx.x * 64;

  v8f acc[2][4];
#pragma unroll
  for (int ms = 0; ms < 2; ++ms)
#pragma unroll
    for (int f = 0; f < 4; ++f)
#pragma unroll
      for (int e = 0; e < 8; ++e) acc[ms][f][e] = 0.f;

  const __bf16* ab = attnT + (size_t)b * L_ * C_;
  for (int kc = 0; kc < C_; kc += 32) {
    __builtin_prefetch(wo + (size_t)o0 * C_ + kc + 64, 0, 3);
    v16bf a0 = load_a(wo + (size_t)(o0)      * C_ + kc, C_);
    v16bf a1 = load_a(wo + (size_t)(o0 + 16) * C_ + kc, C_);
#pragma unroll
    for (int f = 0; f < 4; ++f) {
      v16bf bf = load_bt(ab + (size_t)(l0 + 16 * f) * C_ + kc, C_);
      acc[0][f] = wmma_bf16(a0, bf, acc[0][f]);
      acc[1][f] = wmma_bf16(a1, bf, acc[1][f]);
    }
  }

#pragma unroll
  for (int ms = 0; ms < 2; ++ms) {
#pragma unroll
    for (int f = 0; f < 4; ++f) {
#pragma unroll
      for (int r = 0; r < 8; ++r) {
        int o = o0 + 16 * ms + r + 8 * hl;
        int l = l0 + 16 * f + n;
        out[((size_t)b * C_ + o) * L_ + l] = acc[ms][f][r] + bias[o];
      }
    }
  }
}

// ---------------------------------------------------------------------------
extern "C" void kernel_launch(void* const* d_in, const int* in_sizes, int n_in,
                              void* d_out, int out_size, void* d_ws, size_t ws_size,
                              hipStream_t stream) {
  const float* x    = (const float*)d_in[0];   // [4,512,2048]
  const float* wqkv = (const float*)d_in[1];   // [1536,512]
  const float* wout = (const float*)d_in[2];   // [512,512]
  const float* bout = (const float*)d_in[3];   // [512]
  float* out = (float*)d_out;                  // [4,512,2048]

  char* ws = (char*)d_ws;
  const size_t MB = 1024ull * 1024ull;
  __bf16* xT    = (__bf16*)(ws);               // 8   MB: [b,l,c]
  __bf16* wq_b  = (__bf16*)(ws + 8 * MB);      // 1.5 MB
  __bf16* wo_b  = (__bf16*)(ws + 10 * MB);     // 0.5 MB
  __bf16* qs    = (__bf16*)(ws + 11 * MB);     // 8 MB: [b,h,l,d]
  __bf16* ks    = (__bf16*)(ws + 19 * MB);     // 8 MB
  __bf16* vs    = (__bf16*)(ws + 27 * MB);     // 8 MB
  __bf16* attnT = (__bf16*)(ws + 35 * MB);     // 8 MB: [b,l,c]

  convert_weights<<<(O3_ * C_ + 255) / 256, 256, 0, stream>>>(wqkv, wout, wq_b, wo_b);
  convert_xT<<<(B_ * C_ * L_ + 255) / 256, 256, 0, stream>>>(x, xT);
  qkv_gemm<<<dim3(L_ / 64, O3_ / 256, B_), 256, 0, stream>>>(wq_b, xT, qs, ks, vs);
  attn_kernel<<<dim3(L_ / 64, B_ * H_), 128, 0, stream>>>(qs, ks, vs, attnT);
  out_gemm<<<dim3(L_ / 64, C_ / 256, B_), 256, 0, stream>>>(wo_b, attnT, bout, out);
}
